// VisionTransformer_22187801051602
// MI455X (gfx1250) — matrix operand order repaired and checked by reading, not compiled, not run on hardware
//
#include <hip/hip_runtime.h>

// ---------------- problem constants ----------------
#define BATCH   256
#define T1      65
#define DMODEL  256
#define NHEAD   8
#define HDIM    32
#define MLP_H   2048
#define HEAD_H  8192
#define NCLS    10
#define MROWS   (BATCH * T1)       // 16640, divisible by 64
#define FLATD   (T1 * DMODEL)      // 16640

typedef __attribute__((ext_vector_type(16))) __bf16 v16bf;
typedef __attribute__((ext_vector_type(8)))  float  v8f;

union Frag { v16bf v; unsigned u[8]; uint4 q[2]; };

__device__ __forceinline__ unsigned short f2bf(float f) {
    unsigned u = __float_as_uint(f);
    unsigned r = u + 0x7FFFu + ((u >> 16) & 1u);   // round-to-nearest-even
    return (unsigned short)(r >> 16);
}
__device__ __forceinline__ float bf2f(unsigned short h) {
    return __uint_as_float(((unsigned)h) << 16);
}

// ---------------- fp32 -> bf16 conversion ----------------
__global__ void __launch_bounds__(256) k_convert_bf16(const float* __restrict__ src,
                                                      unsigned short* __restrict__ dst,
                                                      long n) {
    long i = (long)blockIdx.x * 256 + threadIdx.x;
    if (i < n) dst[i] = f2bf(src[i]);
}

// ---------------- patch embed + cls + positional encoding ----------------
__global__ void __launch_bounds__(256) k_patch_embed(const float* __restrict__ x,
                                                     const float* __restrict__ cls,
                                                     const float* __restrict__ Wp,
                                                     const float* __restrict__ bp,
                                                     float* __restrict__ X) {
    long idx = (long)blockIdx.x * 256 + threadIdx.x;  // over B*T1*D
    if (idx >= (long)BATCH * T1 * DMODEL) return;
    int d = (int)(idx % DMODEL);
    long bt = idx / DMODEL;
    int t = (int)(bt % T1);
    int b = (int)(bt / T1);

    float acc;
    if (t == 0) {
        acc = cls[d];
    } else {
        int p  = t - 1;
        int ti = p >> 3, tj = p & 7;           // 8x8 patch grid
        acc = bp[d];
        #pragma unroll
        for (int c = 0; c < 3; ++c)
            #pragma unroll
            for (int py = 0; py < 4; ++py)
                #pragma unroll
                for (int px = 0; px < 4; ++px) {
                    float xv = x[(((size_t)b * 3 + c) * 32 + ti * 4 + py) * 32 + tj * 4 + px];
                    acc += xv * Wp[(size_t)(c * 16 + py * 4 + px) * DMODEL + d];
                }
    }
    // sinusoidal positional encoding: pairs (sin, cos)
    int j = d >> 1;
    float div = __expf(-(float)(2 * j) * (9.210340371976184f / (float)DMODEL));
    float ang = (float)t * div;
    acc += (d & 1) ? __cosf(ang) : __sinf(ang);
    X[idx] = acc;
}

// ---------------- LayerNorm over (T1,D) per sample, bf16 out ----------------
__global__ void __launch_bounds__(256) k_ln2d(const float* __restrict__ X,
                                              const float* __restrict__ w,
                                              const float* __restrict__ bp,
                                              unsigned short* __restrict__ out) {
    __shared__ float s1[256], s2[256];
    int b = blockIdx.x, tid = threadIdx.x;
    const float* xb = X + (size_t)b * FLATD;
    float s = 0.f, sq = 0.f;
    for (int i = tid; i < FLATD; i += 256) { float v = xb[i]; s += v; sq += v * v; }
    s1[tid] = s; s2[tid] = sq; __syncthreads();
    for (int o = 128; o > 0; o >>= 1) {
        if (tid < o) { s1[tid] += s1[tid + o]; s2[tid] += s2[tid + o]; }
        __syncthreads();
    }
    float mu  = s1[0] * (1.0f / FLATD);
    float var = s2[0] * (1.0f / FLATD) - mu * mu;
    float rs  = rsqrtf(var + 1e-5f);
    for (int i = tid; i < FLATD; i += 256) {
        float v = (xb[i] - mu) * rs * w[i] + bp[i];
        out[(size_t)b * FLATD + i] = f2bf(v);
    }
}

// ---------------- generic bf16 WMMA GEMM (compile-time N,K) ----------------
// C[M,N] = act( A[M,K] @ B[K,N] + bias [+ residual] ), A/B bf16 row-major.
// Block: 256 threads = 8 waves; block tile 64(M) x 256(N); each wave owns
// 64x32 (4 M-tiles x 2 N-tiles => 8 v_wmma per K-step of 32).
// Double-buffered LDS: A tile (64x32) via global_load_async_to_lds_b128
// (ASYNCcnt), B tile (32x256) staged TRANSPOSED via uint4 loads + v_perm_b32.
// Next tile's staging is issued before the current tile's WMMAs.
// Requires M%64==0, N%256==0, K%32==0 (true here); EXEC all-ones for WMMA.
template<int N, int K, int ACT, bool HAS_RES, bool OUT_F32, bool OUT_BF16>
__global__ void __launch_bounds__(256) k_gemm_bf16(const unsigned short* __restrict__ A,
                                                   const unsigned short* __restrict__ B,
                                                   const float* __restrict__ bias,
                                                   const float* __restrict__ residual,
                                                   float* __restrict__ Cf,
                                                   unsigned short* __restrict__ Cbf,
                                                   int M) {
    __shared__ unsigned short a_s[2][64 * 32];     // [m][k] row-major
    __shared__ unsigned short b_t[2][256 * 32];    // [n][k] transposed

    const int tid   = threadIdx.x;
    const int lane  = tid & 31;
    const int wave  = tid >> 5;
    const int m0    = blockIdx.y * 64;
    const int nb    = blockIdx.x * 256;
    const int col   = lane & 15;
    const int halfs = lane >> 4;

    const int arow = tid >> 2;          // 0..63
    const int ac4  = tid & 3;           // 16-byte chunk within row (32 bf16 = 4 chunks)
    const unsigned a_lds0 = (unsigned)(size_t)&a_s[0][arow * 32 + ac4 * 8];
    const unsigned a_lds1 = (unsigned)(size_t)&a_s[1][arow * 32 + ac4 * 8];

    // stage K-step `kkv` into LDS buffer `bufv`
    auto stage = [&](int kkv, int bufv) {
        // A tile: 64 rows x 64B, one async b128 per thread (layout-preserving copy)
        unsigned goff = (unsigned)(((size_t)(m0 + arow) * K + kkv + ac4 * 8) * 2);
        asm volatile("global_load_async_to_lds_b128 %0, %1, %2"
                     :: "v"(bufv ? a_lds1 : a_lds0), "v"(goff), "s"(A) : "memory");
        // B tile transposed: 512 work items = (16 k-pairs) x (32 col-groups of 8)
        unsigned* bt = (unsigned*)b_t[bufv];
        #pragma unroll
        for (int i = 0; i < 2; ++i) {
            int e  = tid + i * 256;
            int kp = e >> 5;                // k-pair 0..15
            int g8 = e & 31;                // column group of 8
            uint4 lo4 = *(const uint4*)(B + (size_t)(kkv + 2 * kp)     * N + nb + g8 * 8);
            uint4 hi4 = *(const uint4*)(B + (size_t)(kkv + 2 * kp + 1) * N + nb + g8 * 8);
            unsigned lw[4] = {lo4.x, lo4.y, lo4.z, lo4.w};
            unsigned hw[4] = {hi4.x, hi4.y, hi4.z, hi4.w};
            #pragma unroll
            for (int j = 0; j < 4; ++j) {
                unsigned w0 = __builtin_amdgcn_perm(hw[j], lw[j], 0x05040100u);
                unsigned w1 = __builtin_amdgcn_perm(hw[j], lw[j], 0x07060302u);
                int n = g8 * 8 + 2 * j;
                bt[(n)     * 16 + kp] = w0;
                bt[(n + 1) * 16 + kp] = w1;
            }
        }
    };

    v8f zero = {0.f, 0.f, 0.f, 0.f, 0.f, 0.f, 0.f, 0.f};
    v8f acc[4][2];
    #pragma unroll
    for (int mt = 0; mt < 4; ++mt) { acc[mt][0] = zero; acc[mt][1] = zero; }

    constexpr int steps = K / 32;
    stage(0, 0);
    asm volatile("s_wait_asynccnt 0" ::: "memory");
    __syncthreads();

    for (int it = 0; it < steps; ++it) {
        const int buf = it & 1;
        // --- fragments from current buffer (all b128 LDS loads) ---
        Frag a[4], bfr[2];
        #pragma unroll
        for (int mt = 0; mt < 4; ++mt) {
            int m = mt * 16 + col;
            a[mt].q[0] = *(const uint4*)&a_s[buf][m * 32 + halfs * 8];
            a[mt].q[1] = *(const uint4*)&a_s[buf][m * 32 + 16 + halfs * 8];
        }
        #pragma unroll
        for (int nt = 0; nt < 2; ++nt) {
            int n = wave * 32 + nt * 16 + col;
            bfr[nt].q[0] = *(const uint4*)&b_t[buf][n * 32 + halfs * 8];
            bfr[nt].q[1] = *(const uint4*)&b_t[buf][n * 32 + 16 + halfs * 8];
        }
        // --- prefetch next K-step into other buffer ---
        if (it + 1 < steps) stage((it + 1) * 32, buf ^ 1);
        // --- 8 WMMAs ---
        #pragma unroll
        for (int mt = 0; mt < 4; ++mt)
            #pragma unroll
            for (int nt = 0; nt < 2; ++nt)
                acc[mt][nt] = __builtin_amdgcn_wmma_f32_16x16x32_bf16(
                    false, a[mt].v, false, bfr[nt].v, (short)0, acc[mt][nt],
                    false, false);
        if (it + 1 < steps) {
            asm volatile("s_wait_asynccnt 0" ::: "memory");
            __syncthreads();
        }
    }

    // --- epilogue: C layout VGPR r -> (M = r + halfs*8, N = lane%16) ---
    #pragma unroll
    for (int mt = 0; mt < 4; ++mt) {
        #pragma unroll
        for (int nt = 0; nt < 2; ++nt) {
            int gn = nb + wave * 32 + nt * 16 + col;
            #pragma unroll
            for (int r = 0; r < 8; ++r) {
                int gm = m0 + mt * 16 + r + halfs * 8;
                float v = acc[mt][nt][r] + bias[gn];
                if constexpr (HAS_RES) v += residual[(size_t)gm * N + gn];
                if constexpr (ACT == 1) v = fmaxf(v, 0.f);                         // relu
                if constexpr (ACT == 2) v = v * __fdividef(1.f, 1.f + __expf(-v)); // silu
                if constexpr (OUT_F32)  Cf [(size_t)gm * N + gn] = v;
                if constexpr (OUT_BF16) Cbf[(size_t)gm * N + gn] = f2bf(v);
            }
        }
    }
}

// ---------------- attention per (b,h), fused residual ----------------
__global__ void __launch_bounds__(96) k_attention(const float* __restrict__ qkv,
                                                  const float* __restrict__ X,
                                                  float* __restrict__ R) {
    int bh = blockIdx.x;
    int b = bh >> 3, h = bh & 7;
    __shared__ float ks[T1 * HDIM], vs[T1 * HDIM];
    const float* base = qkv + (size_t)b * T1 * (3 * DMODEL);
    for (int i = threadIdx.x; i < T1 * HDIM; i += 96) {
        int t = i >> 5, hd = i & 31;
        ks[i] = base[(size_t)t * 768 + h * 96 + 32 + hd];
        vs[i] = base[(size_t)t * 768 + h * 96 + 64 + hd];
    }
    __syncthreads();
    int t = threadIdx.x;
    if (t < T1) {
        float q[HDIM];
        #pragma unroll
        for (int hd = 0; hd < HDIM; ++hd) q[hd] = base[(size_t)t * 768 + h * 96 + hd];
        const float scale = 0.17677669529663687f;  // 1/sqrt(32)
        float mx = -1e30f;
        for (int j = 0; j < T1; ++j) {
            float s = 0.f;
            #pragma unroll
            for (int hd = 0; hd < HDIM; ++hd) s += q[hd] * ks[j * 32 + hd];
            mx = fmaxf(mx, s * scale);
        }
        float sum = 0.f, out[HDIM];
        #pragma unroll
        for (int hd = 0; hd < HDIM; ++hd) out[hd] = 0.f;
        for (int j = 0; j < T1; ++j) {
            float s = 0.f;
            #pragma unroll
            for (int hd = 0; hd < HDIM; ++hd) s += q[hd] * ks[j * 32 + hd];
            float e = __expf(s * scale - mx);
            sum += e;
            #pragma unroll
            for (int hd = 0; hd < HDIM; ++hd) out[hd] += e * vs[j * 32 + hd];
        }
        float inv = 1.f / sum;
        #pragma unroll
        for (int hd = 0; hd < HDIM; ++hd) {
            size_t o = (size_t)b * T1 * DMODEL + (size_t)t * DMODEL + h * 32 + hd;
            R[o] = out[hd] * inv + X[o];
        }
    }
}

// ---------------- tiny classifier head2 ----------------
__global__ void __launch_bounds__(256) k_head2(const unsigned short* __restrict__ hidden,
                                               const float* __restrict__ W,
                                               const float* __restrict__ bias,
                                               float* __restrict__ out) {
    __shared__ float red[256];
    int b = blockIdx.x, tid = threadIdx.x;
    const unsigned short* hb = hidden + (size_t)b * HEAD_H;
    float acc[NCLS];
    #pragma unroll
    for (int c = 0; c < NCLS; ++c) acc[c] = 0.f;
    for (int k = tid; k < HEAD_H; k += 256) {
        float hv = bf2f(hb[k]);
        const float* wr = W + (size_t)k * NCLS;
        #pragma unroll
        for (int c = 0; c < NCLS; ++c) acc[c] += hv * wr[c];
    }
    for (int c = 0; c < NCLS; ++c) {
        red[tid] = acc[c]; __syncthreads();
        for (int o = 128; o > 0; o >>= 1) {
            if (tid < o) red[tid] += red[tid + o];
            __syncthreads();
        }
        if (tid == 0) out[b * NCLS + c] = red[0] + bias[c];
        __syncthreads();
    }
}

// ---------------- host driver ----------------
extern "C" void kernel_launch(void* const* d_in, const int* in_sizes, int n_in,
                              void* d_out, int out_size, void* d_ws, size_t ws_size,
                              hipStream_t stream) {
    const float* x       = (const float*)d_in[0];
    const float* cls     = (const float*)d_in[1];
    const float* Wp      = (const float*)d_in[2];
    const float* bp      = (const float*)d_in[3];
    const float* qkv_w   = (const float*)d_in[4];
    const float* qkv_b   = (const float*)d_in[5];
    const float* ln_w    = (const float*)d_in[6];
    const float* ln_b    = (const float*)d_in[7];
    const float* mlp1_w  = (const float*)d_in[8];
    const float* mlp1_b  = (const float*)d_in[9];
    const float* mlp2_w  = (const float*)d_in[10];
    const float* mlp2_b  = (const float*)d_in[11];
    const float* head1_w = (const float*)d_in[12];
    const float* head1_b = (const float*)d_in[13];
    const float* head2_w = (const float*)d_in[14];
    const float* head2_b = (const float*)d_in[15];

    char* ws = (char*)d_ws;
    size_t off = 0;
    auto alloc = [&](size_t bytes) -> void* {
        off = (off + 255) & ~(size_t)255;
        void* p = ws + off;
        off += bytes;
        return p;
    };

    float*          Xf    = (float*)alloc((size_t)MROWS * DMODEL * 4);
    float*          Rf    = (float*)alloc((size_t)MROWS * DMODEL * 4);
    unsigned short* Hbf   = (unsigned short*)alloc((size_t)MROWS * DMODEL * 2);
    float*          QKVf  = (float*)alloc((size_t)MROWS * 3 * DMODEL * 4);
    unsigned short* Mbf   = (unsigned short*)alloc((size_t)MROWS * MLP_H * 2);
    unsigned short* Hidbf = (unsigned short*)alloc((size_t)BATCH * HEAD_H * 2);
    unsigned short* QKVWb = (unsigned short*)alloc((size_t)DMODEL * 3 * DMODEL * 2);
    unsigned short* M1Wb  = (unsigned short*)alloc((size_t)DMODEL * MLP_H * 2);
    unsigned short* M2Wb  = (unsigned short*)alloc((size_t)MLP_H * DMODEL * 2);
    unsigned short* H1Wb  = (unsigned short*)alloc((size_t)FLATD * HEAD_H * 2);

    // 1) patch embed + cls + PE
    {
        long n = (long)BATCH * T1 * DMODEL;
        k_patch_embed<<<dim3((unsigned)((n + 255) / 256)), 256, 0, stream>>>(x, cls, Wp, bp, Xf);
    }

    // 2) two transformer layers
    for (int i = 0; i < 2; ++i) {
        const float* lnw = ln_w + (size_t)i * FLATD;
        const float* lnb = ln_b + (size_t)i * FLATD;

        // convert weights for this layer
        {
            long n = (long)DMODEL * 3 * DMODEL;
            k_convert_bf16<<<dim3((unsigned)((n + 255) / 256)), 256, 0, stream>>>(
                qkv_w + (size_t)i * n, QKVWb, n);
            long n1 = (long)DMODEL * MLP_H;
            k_convert_bf16<<<dim3((unsigned)((n1 + 255) / 256)), 256, 0, stream>>>(
                mlp1_w + (size_t)i * n1, M1Wb, n1);
            long n2 = (long)MLP_H * DMODEL;
            k_convert_bf16<<<dim3((unsigned)((n2 + 255) / 256)), 256, 0, stream>>>(
                mlp2_w + (size_t)i * n2, M2Wb, n2);
        }

        // LN(X) -> Hbf
        k_ln2d<<<BATCH, 256, 0, stream>>>(Xf, lnw, lnb, Hbf);
        // QKV = Hbf @ qkv_w + qkv_b        [16640 x 768 x 256]
        k_gemm_bf16<768, 256, 0, false, true, false>
            <<<dim3(768 / 256, MROWS / 64), 256, 0, stream>>>(
                Hbf, QKVWb, qkv_b + (size_t)i * 3 * DMODEL, nullptr,
                QKVf, nullptr, MROWS);
        // attention + residual -> Rf
        k_attention<<<BATCH * NHEAD, 96, 0, stream>>>(QKVf, Xf, Rf);
        // LN(R) -> Hbf
        k_ln2d<<<BATCH, 256, 0, stream>>>(Rf, lnw, lnb, Hbf);
        // MLP1: relu(Hbf @ W1 + b1) -> Mbf [16640 x 2048 x 256]
        k_gemm_bf16<MLP_H, 256, 1, false, false, true>
            <<<dim3(MLP_H / 256, MROWS / 64), 256, 0, stream>>>(
                Hbf, M1Wb, mlp1_b + (size_t)i * MLP_H, nullptr,
                nullptr, Mbf, MROWS);
        // MLP2: Mbf @ W2 + b2 + R -> Xf    [16640 x 256 x 2048]
        k_gemm_bf16<DMODEL, MLP_H, 0, true, true, false>
            <<<dim3(DMODEL / 256, MROWS / 64), 256, 0, stream>>>(
                Mbf, M2Wb, mlp2_b + (size_t)i * DMODEL, Rf,
                Xf, nullptr, MROWS);
    }

    // 3) classifier head
    {
        long n = (long)MROWS * DMODEL;  // final activations -> bf16 (viewed [256, 16640])
        k_convert_bf16<<<dim3((unsigned)((n + 255) / 256)), 256, 0, stream>>>(Xf, Hbf, n);
        long nw = (long)FLATD * HEAD_H;
        k_convert_bf16<<<dim3((unsigned)((nw + 255) / 256)), 256, 0, stream>>>(head1_w, H1Wb, nw);
        // hidden = silu(flat @ head1_w + b) [256 x 8192 x 16640]
        k_gemm_bf16<HEAD_H, FLATD, 2, false, false, true>
            <<<dim3(HEAD_H / 256, BATCH / 64), 256, 0, stream>>>(
                Hbf, H1Wb, head1_b, nullptr, nullptr, Hidbf, BATCH);
        // logits
        k_head2<<<BATCH, 256, 0, stream>>>(Hidbf, head2_w, head2_b, (float*)d_out);
    }
}
